// CommonsenseattentionCell_86749749444652
// MI455X (gfx1250) — matrix operand order repaired and checked by reading, not compiled, not run on hardware
//
#include <hip/hip_runtime.h>
#include <math.h>

typedef __bf16 bf16;
typedef __attribute__((ext_vector_type(16))) __bf16 v16bf;
typedef __attribute__((ext_vector_type(8)))  float  v8f;

#define BM 128
#define BN 128
#define BK 32
#define LDT 36   // LDS K-stride in halves: 72B row = odd multiple of 8B -> b64-friendly, bank-conflict free

// pack two f32 into packed bf16 pair (RNE), staying in 32-bit domain:
// 2x (bfe + add3) + 1x v_perm_b32  ->  ~2.5 VALU per element
__device__ __forceinline__ unsigned pack2bf(float lo, float hi) {
    union { float f; unsigned u; } cl, ch;
    cl.f = lo; ch.f = hi;
    unsigned ul = cl.u + (0x7fffu + ((cl.u >> 16) & 1u));
    unsigned uh = ch.u + (0x7fffu + ((ch.u >> 16) & 1u));
    // dst = { uh.b3, uh.b2, ul.b3, ul.b2 } = hi16(uh):hi16(ul)
    return __builtin_amdgcn_perm(uh, ul, 0x07060302u);
}

// Y[b] = (SCALE? rowScale[b] : 1) * X_row(b) @ W_g^T + bias_g
// GROUPED: group s = blockIdx.z, rows gathered via rowList/grpCnt, weights at W + s*N*K
template<bool GROUPED, bool SCALE>
__global__ void __launch_bounds__(256)
gemm_wmma(const float* __restrict__ X, long xStride, long xGroupOff,
          const float* __restrict__ W, const float* __restrict__ bias,
          float* __restrict__ Y, long yStride,
          const float* __restrict__ rowScale,
          const int* __restrict__ rowList,
          const int* __restrict__ grpCnt,
          int M, int N, int K, int maxM)
{
    __shared__ unsigned short As[2][BM][LDT];
    __shared__ unsigned short Bs[2][BN][LDT];

    const int s    = GROUPED ? blockIdx.z : 0;
    const int tid  = threadIdx.x;
    const int lane = tid & 31;
    const int wv   = tid >> 5;
    const int wm   = wv & 3;      // 4 waves along M, 32 rows each
    const int wn   = wv >> 2;     // 2 waves along N, 64 cols each

    int Meff = M;
    const int* rl = nullptr;
    if (GROUPED) { Meff = grpCnt[s]; rl = rowList + (size_t)s * maxM; }

    const int rowBase = blockIdx.y * BM;
    if (rowBase >= Meff) return;
    const int nBase = blockIdx.x * BN;

    const float* Wg = W    + (size_t)s * (size_t)N * (size_t)K;
    const float* bg = bias + (size_t)s * N;
    const long  xso = GROUPED ? (long)s * xGroupOff : 0L;

    // ---- per-thread staging geometry: 4 float4 loads each for A and B ----
    // linear f4 id = tid + i*256 -> row = (tid>>3)+i*32, col = (tid&7)*4 (same for all i)
    const int cOff = (tid & 7) * 4;
    const float* aPtr[4];
    const float* bPtr[4];
    float aScl[4];
    bool  aOk[4];
    #pragma unroll
    for (int i = 0; i < 4; ++i) {
        int r  = (tid >> 3) + i * 32;
        int gr = rowBase + r;
        aOk[i] = gr < Meff;
        int b  = aOk[i] ? (GROUPED ? rl[gr] : gr) : 0;
        aPtr[i] = X + (long)b * xStride + xso;
        aScl[i] = SCALE ? rowScale[b] : 1.0f;
        bPtr[i] = Wg + (size_t)(nBase + r) * K;
    }

    v8f acc[2][4];
    #pragma unroll
    for (int i = 0; i < 2; ++i)
        #pragma unroll
        for (int j = 0; j < 4; ++j)
            #pragma unroll
            for (int r = 0; r < 8; ++r) acc[i][j][r] = 0.0f;

    float4 aReg[4], bReg[4];
    auto loadRegs = [&](int k0) {
        #pragma unroll
        for (int i = 0; i < 4; ++i) {
            if (aOk[i]) aReg[i] = *(const float4*)(aPtr[i] + k0 + cOff);
            else        aReg[i] = make_float4(0.f, 0.f, 0.f, 0.f);
            bReg[i] = *(const float4*)(bPtr[i] + k0 + cOff);
        }
    };
    auto storeRegs = [&](int buf) {
        #pragma unroll
        for (int i = 0; i < 4; ++i) {
            int r = (tid >> 3) + i * 32;
            float4 a = aReg[i];
            if (SCALE) { a.x *= aScl[i]; a.y *= aScl[i]; a.z *= aScl[i]; a.w *= aScl[i]; }
            uint2 pa; pa.x = pack2bf(a.x, a.y); pa.y = pack2bf(a.z, a.w);
            *(uint2*)&As[buf][r][cOff] = pa;
            float4 b = bReg[i];
            uint2 pb; pb.x = pack2bf(b.x, b.y); pb.y = pack2bf(b.z, b.w);
            *(uint2*)&Bs[buf][r][cOff] = pb;
        }
    };

    loadRegs(0);
    int buf = 0;
    for (int k0 = 0; k0 < K; k0 += BK) {
        storeRegs(buf);
        __syncthreads();
        if (k0 + BK < K) loadRegs(k0 + BK);   // next tile's global loads overlap WMMA phase

        // A fragments (16x32 bf16): lane->M, halves->K (hi lanes K+8)
        v16bf afrag[2];
        const int koffA = (lane >= 16) ? 8 : 0;
        #pragma unroll
        for (int i = 0; i < 2; ++i) {
            int mrow = wm * 32 + i * 16 + (lane & 15);
            const unsigned short* ab = &As[buf][mrow][0];
            union { v16bf v; unsigned u[8]; } au;
            #pragma unroll
            for (int v = 0; v < 8; ++v) {
                int kk = (v < 4) ? (2 * v + koffA) : (16 + 2 * (v - 4) + koffA);
                au.u[v] = *(const unsigned*)(ab + kk);
            }
            afrag[i] = au.v;
        }
        // B fragments (32x16 bf16): lane->N, halves->K (hi lanes K+8)
        const int koffB = (lane >= 16) ? 8 : 0;
        #pragma unroll
        for (int j = 0; j < 4; ++j) {
            int nrow = wn * 64 + j * 16 + (lane & 15);
            const unsigned short* bb = &Bs[buf][nrow][koffB];
            union { v16bf v; unsigned u[8]; } bu;
            #pragma unroll
            for (int h = 0; h < 8; ++h) bu.u[h] = *(const unsigned*)(bb + ((h < 4) ? 2 * h : (16 + 2 * (h - 4))));
            #pragma unroll
            for (int i = 0; i < 2; ++i)
                acc[i][j] = __builtin_amdgcn_wmma_f32_16x16x32_bf16(
                    false, afrag[i], false, bu.v, (short)0, acc[i][j], false, false);
        }
        buf ^= 1;
    }

    // epilogue: C layout VGPR r -> M = r + (lane>=16?8:0), lane&15 -> N
    const int mhi = (lane >= 16) ? 8 : 0;
    #pragma unroll
    for (int i = 0; i < 2; ++i) {
        #pragma unroll
        for (int j = 0; j < 4; ++j) {
            int gcol = nBase + wn * 64 + j * 16 + (lane & 15);
            float bv = bg[gcol];
            #pragma unroll
            for (int r = 0; r < 8; ++r) {
                int gm = rowBase + wm * 32 + i * 16 + mhi + r;
                if (gm < Meff) {
                    int b = GROUPED ? rl[gm] : gm;
                    Y[(long)b * yStride + gcol] = acc[i][j][r] + bv;
                }
            }
        }
    }
}

__global__ void dot_rows(const float* __restrict__ qx, const float* __restrict__ kx,
                         float* __restrict__ score, int D)
{
    __shared__ float red[256];
    int b = blockIdx.x;
    float acc = 0.0f;
    for (int i = threadIdx.x; i < D; i += 256)
        acc += qx[(size_t)b * D + i] * kx[(size_t)b * D + i];
    red[threadIdx.x] = acc;
    __syncthreads();
    for (int s = 128; s > 0; s >>= 1) {
        if ((int)threadIdx.x < s) red[threadIdx.x] += red[threadIdx.x + s];
        __syncthreads();
    }
    if (threadIdx.x == 0) score[b] = red[0];
}

// softmax over the batch axis (B values), single 1024-thread block
__global__ void softmax_col(const float* __restrict__ score, float* __restrict__ sm, int B)
{
    __shared__ float red[1024];
    int t = threadIdx.x;
    float v = (t < B) ? score[t] : -3.402823466e38f;
    red[t] = v;
    __syncthreads();
    for (int s = 512; s > 0; s >>= 1) {
        if (t < s) red[t] = fmaxf(red[t], red[t + s]);
        __syncthreads();
    }
    float mx = red[0];
    __syncthreads();
    float e = (t < B) ? __expf(v - mx) : 0.0f;
    red[t] = e;
    __syncthreads();
    for (int s = 512; s > 0; s >>= 1) {
        if (t < s) red[t] += red[t + s];
        __syncthreads();
    }
    if (t < B) sm[t] = e / red[0];
}

// argmax of one-hot masks + per-slot compaction lists (one 1024-thread block)
__global__ void build_index(const float* __restrict__ qm, const float* __restrict__ qmn,
                            int* __restrict__ idxS, int* __restrict__ idxL,
                            int* __restrict__ cntS, int* __restrict__ listS,
                            int* __restrict__ cntL, int* __restrict__ listL,
                            int B, int S)
{
    __shared__ int lc[32];
    int t = threadIdx.x;
    if (t < 2 * S) lc[t] = 0;
    __syncthreads();
    if (t < B) {
        int is = 0; float bv = qm[(size_t)t * S];
        for (int s = 1; s < S; ++s) { float v = qm[(size_t)t * S + s]; if (v > bv) { bv = v; is = s; } }
        idxS[t] = is;
        int p = atomicAdd(&lc[is], 1);
        listS[is * B + p] = t;

        int il = 0; float bl = qmn[(size_t)t * S];
        for (int s = 1; s < S; ++s) { float v = qmn[(size_t)t * S + s]; if (v > bl) { bl = v; il = s; } }
        idxL[t] = il;
        int q = atomicAdd(&lc[S + il], 1);
        listL[il * B + q] = t;
    }
    __syncthreads();
    if (t < S) { cntS[t] = lc[t]; cntL[t] = lc[S + t]; }
}

__global__ void gru_elem(const float* __restrict__ gi, const float* __restrict__ gh,
                         const float* __restrict__ s0, const int* __restrict__ idx,
                         float* __restrict__ out, int B, int S, int H)
{
    int i = blockIdx.x * blockDim.x + threadIdx.x;
    if (i >= B * H) return;
    int b = i / H, j = i % H;
    const float* gib = gi + (size_t)b * 3 * H;
    const float* ghb = gh + (size_t)b * 3 * H;
    float r = 1.0f / (1.0f + __expf(-(gib[j] + ghb[j])));
    float z = 1.0f / (1.0f + __expf(-(gib[H + j] + ghb[H + j])));
    float n = tanhf(gib[2 * H + j] + r * ghb[2 * H + j]);
    float h = s0[((size_t)b * S + idx[b]) * H + j];
    out[i] = (1.0f - z) * n + z * h;
}

__global__ void assemble(const float* __restrict__ s0,
                         const float* __restrict__ snew, const float* __restrict__ lnew,
                         const int* __restrict__ idxS, const int* __restrict__ idxL,
                         float* __restrict__ out, int B, int S, int H)
{
    int i = blockIdx.x * blockDim.x + threadIdx.x;
    if (i >= B * S * H) return;
    int h = i % H;
    int t = i / H;
    int s = t % S;
    int b = t / S;
    float v = s0[i];
    if (s == idxS[b])      v = snew[(size_t)b * H + h];
    else if (s == idxL[b]) v = lnew[(size_t)b * H + h];
    out[i] = v;
}

extern "C" void kernel_launch(void* const* d_in, const int* in_sizes, int n_in,
                              void* d_out, int out_size, void* d_ws, size_t ws_size,
                              hipStream_t stream)
{
    (void)in_sizes; (void)out_size; (void)ws_size;
    const int B = 1024, D = 1024, H = 1024, S = 16;

    const float* u   = (const float*)d_in[0];
    const float* sk  = (const float*)d_in[1];
    const float* nu  = (const float*)d_in[2];
    const float* lk  = (const float*)d_in[3];
    const float* s0  = (const float*)d_in[5];
    const float* qm  = (const float*)d_in[6];
    const float* qmn = (const float*)d_in[7];

    struct AttP { const float *Wk, *bk, *Wq, *bq, *Wp, *bp; };
    AttP att[4];
    const float *Wih, *Whh, *bih, *bhh;
    if (n_in >= 36) {
        // pytree-flattened: each att tuple contributes 6 leaves
        for (int a = 0; a < 4; ++a) {
            int base = 8 + a * 6;
            att[a] = { (const float*)d_in[base],     (const float*)d_in[base + 1],
                       (const float*)d_in[base + 2], (const float*)d_in[base + 3],
                       (const float*)d_in[base + 4], (const float*)d_in[base + 5] };
        }
        Wih = (const float*)d_in[32]; Whh = (const float*)d_in[33];
        bih = (const float*)d_in[34]; bhh = (const float*)d_in[35];
    } else {
        // each att tuple as one flat blob: Wk|bk|Wq|bq|Wp|bp
        for (int a = 0; a < 4; ++a) {
            const float* p = (const float*)d_in[8 + a];
            att[a].Wk = p; p += (size_t)D * D;
            att[a].bk = p; p += D;
            att[a].Wq = p; p += (size_t)D * D;
            att[a].bq = p; p += D;
            att[a].Wp = p; p += (size_t)D * D;
            att[a].bp = p;
        }
        Wih = (const float*)d_in[12]; Whh = (const float*)d_in[13];
        bih = (const float*)d_in[14]; bhh = (const float*)d_in[15];
    }

    // workspace layout
    float* wsf = (float*)d_ws;
    const size_t MD = (size_t)B * D;
    float* kx    = wsf; wsf += MD;
    float* qx    = wsf; wsf += MD;
    float* tA    = wsf; wsf += MD;            // s_0 / s_1 temp
    float* ssb   = wsf; wsf += MD;            // s_s
    float* slb   = wsf; wsf += MD;            // s_l
    float* snew  = wsf; wsf += (size_t)B * H;
    float* lnew  = wsf; wsf += (size_t)B * H;
    float* gi    = wsf; wsf += (size_t)B * 3 * H;
    float* gh    = wsf; wsf += (size_t)B * 3 * H;
    float* score = wsf; wsf += B;
    float* smv   = wsf; wsf += B;
    int* ip    = (int*)wsf;
    int* idxS  = ip; ip += B;
    int* idxL  = ip; ip += B;
    int* cntS  = ip; ip += S;
    int* cntL  = ip; ip += S;
    int* listS = ip; ip += S * B;
    int* listL = ip; ip += S * B;

    dim3 blk(256);
    dim3 gA(D / BN, B / BM, 1);   // 8 x 8 blocks for the 1024^3 attention GEMMs

    build_index<<<1, 1024, 0, stream>>>(qm, qmn, idxS, idxL, cntS, listS, cntL, listL, B, S);

    auto launch_att = [&](const AttP& p, const float* k2d, const float* q2d, float* out) {
        gemm_wmma<false, false><<<gA, blk, 0, stream>>>(
            k2d, (long)D, 0L, p.Wk, p.bk, kx, (long)D,
            (const float*)nullptr, (const int*)nullptr, (const int*)nullptr, B, D, D, B);
        gemm_wmma<false, false><<<gA, blk, 0, stream>>>(
            q2d, (long)D, 0L, p.Wq, p.bq, qx, (long)D,
            (const float*)nullptr, (const int*)nullptr, (const int*)nullptr, B, D, D, B);
        dot_rows<<<dim3(B), dim3(256), 0, stream>>>(qx, kx, score, D);
        softmax_col<<<1, 1024, 0, stream>>>(score, smv, B);
        gemm_wmma<false, true><<<gA, blk, 0, stream>>>(
            kx, (long)D, 0L, p.Wp, p.bp, out, (long)D,
            (const float*)smv, (const int*)nullptr, (const int*)nullptr, B, D, D, B);
    };

    launch_att(att[0], sk, u,  tA);   // s_0
    launch_att(att[1], sk, tA, ssb);  // s_s
    launch_att(att[2], lk, nu, tA);   // s_1
    launch_att(att[3], lk, tA, slb);  // s_l

    // grouped GRU GEMMs: z = slot index, rows gathered per slot (avoids 16x einsum)
    dim3 gG(3 * H / BN, (B + BM - 1) / BM, S);
    gemm_wmma<true, false><<<gG, blk, 0, stream>>>(
        ssb, (long)H, 0L, Wih, bih, gi, (long)(3 * H),
        (const float*)nullptr, listS, cntS, B, 3 * H, D, B);
    gemm_wmma<true, false><<<gG, blk, 0, stream>>>(
        s0, (long)(S * H), (long)H, Whh, bhh, gh, (long)(3 * H),
        (const float*)nullptr, listS, cntS, B, 3 * H, H, B);
    gru_elem<<<(B * H + 255) / 256, 256, 0, stream>>>(gi, gh, s0, idxS, snew, B, S, H);

    gemm_wmma<true, false><<<gG, blk, 0, stream>>>(
        slb, (long)H, 0L, Wih, bih, gi, (long)(3 * H),
        (const float*)nullptr, listL, cntL, B, 3 * H, D, B);
    gemm_wmma<true, false><<<gG, blk, 0, stream>>>(
        s0, (long)(S * H), (long)H, Whh, bhh, gh, (long)(3 * H),
        (const float*)nullptr, listL, cntL, B, 3 * H, H, B);
    gru_elem<<<(B * H + 255) / 256, 256, 0, stream>>>(gi, gh, s0, idxL, lnew, B, S, H);

    assemble<<<(B * S * H + 255) / 256, 256, 0, stream>>>(s0, snew, lnew, idxS, idxL,
                                                          (float*)d_out, B, S, H);
}
